// RingAttention_37203006718195
// MI455X (gfx1250) — matrix operand order repaired and checked by reading, not compiled, not run on hardware
//
#include <hip/hip_runtime.h>
#include <hip/hip_bf16.h>

// ---------------------------------------------------------------------------
// Banded ("ring") attention for MI455X / gfx1250, bf16 WMMA with f32 accum.
//   B=4, S=4096, E=1024, RADIUS=128 -> band width 257 (tiled as 272/288).
// v5: uniform bases laundered via readfirstlane into addrspace(1) pointers ->
//     global_load saddr+voffset form (no per-iteration v_add_nc_u64, fewer
//     address VGPRs, deeper load clauses).
// Workspace: 4*2MB weights + 3*32MB Q/K/Vt + 32MB Ctx = 136 MB.
// ---------------------------------------------------------------------------

#define EDIM   1024
#define SEQ    4096
#define BATCH  4
#define RAD    128
#define MROWS  (BATCH * SEQ)   // 16384

#define GAS __attribute__((address_space(1)))

typedef __attribute__((ext_vector_type(8)))  float  v8f;
typedef __attribute__((ext_vector_type(8)))  __bf16 v8bf;
typedef __attribute__((ext_vector_type(16))) __bf16 v16bf;

typedef const GAS __bf16 gbf16;
typedef const GAS float  gf32;
typedef const GAS v8bf   gv8bf;
typedef const GAS v8f    gv8f;

template <int N> struct IC { static constexpr int v = N; };

// Force a wave-uniform pointer into SGPRs, keeping global addrspace.
static __device__ __forceinline__ gbf16* uni_bf(const __bf16* p) {
  unsigned long long u = (unsigned long long)p;
  unsigned lo = (unsigned)__builtin_amdgcn_readfirstlane((int)(unsigned)u);
  unsigned hi = (unsigned)__builtin_amdgcn_readfirstlane((int)(unsigned)(u >> 32));
  return (gbf16*)(((unsigned long long)hi << 32) | lo);
}
static __device__ __forceinline__ gf32* uni_f(const float* p) {
  unsigned long long u = (unsigned long long)p;
  unsigned lo = (unsigned)__builtin_amdgcn_readfirstlane((int)(unsigned)u);
  unsigned hi = (unsigned)__builtin_amdgcn_readfirstlane((int)(unsigned)(u >> 32));
  return (gf32*)(((unsigned long long)hi << 32) | lo);
}

static __device__ __forceinline__ v16bf join8(v8bf lo, v8bf hi) {
  v16bf r;
#pragma unroll
  for (int i = 0; i < 8; ++i) { r[i] = lo[i]; r[i + 8] = hi[i]; }
  return r;
}

static __device__ __forceinline__ v8f wmma_bf16(v16bf a, v16bf b, v8f c) {
  // D = A(16x32 bf16) x B(32x16 bf16) + C(16x16 f32)
  return __builtin_amdgcn_wmma_f32_16x16x32_bf16(
      false, a, false, b, (short)0, c, false, false);
}

// A fragment (16x32 MxK), ISA 16-bit A layout:
//   lane L: row M=(L&15); K elems {ko..ko+7, ko+16..ko+23}, ko=(L>>4)*8.
static __device__ __forceinline__ v16bf load_a_bf16(const __bf16* rowp) {
  return join8(*(const v8bf*)rowp, *(const v8bf*)(rowp + 16));
}
static __device__ __forceinline__ v16bf load_a_bf16_g(gbf16* rowp) {
  return join8(*(gv8bf*)rowp, *(gv8bf*)(rowp + 16));
}
static __device__ __forceinline__ v16bf load_a_f32_g(gf32* rowp) {
  v8f x = *(gv8f*)rowp;
  v8f y = *(gv8f*)(rowp + 16);
  v16bf r;
#pragma unroll
  for (int i = 0; i < 8; ++i) { r[i] = (__bf16)x[i]; r[i + 8] = (__bf16)y[i]; }
  return r;
}
// B fragment (32x16 KxN): lane L: col N=(L&15); K elems {ko..ko+7, ko+16..ko+23},
// ko=(L>>4)*8.
static __device__ __forceinline__ v16bf load_b_bf16_g(gbf16* colp) {
  return join8(*(gv8bf*)colp, *(gv8bf*)(colp + 16));
}

// ---------------------------------------------------------------------------
// Kernel 1: f32 -> bf16 conversion (weights)
// ---------------------------------------------------------------------------
__global__ void cvt_f32_bf16(const float* __restrict__ x,
                             __bf16* __restrict__ y, int n) {
  int i = blockIdx.x * blockDim.x + threadIdx.x;
  if (i < n) y[i] = (__bf16)x[i];
}

// ---------------------------------------------------------------------------
// Kernel 2: C[M,N] = A[M,K] * W[N,K]^T + bias   (M=16384, N=K=1024)
//   one wave -> 32x128 C tile (2 m-frags x 8 n-frags, 16 accumulators);
//   block = 8 waves (sharing the 32-row A slab via L1); grid = 512 blocks.
// ---------------------------------------------------------------------------
enum { OUT_ROW_BF16 = 0, OUT_TRANS_BF16 = 1, OUT_ROW_F32 = 2 };

template <bool A_BF16, int OUT_MODE>
__global__ __launch_bounds__(256)
void gemm32x128_nt(const void* __restrict__ Ain, const __bf16* __restrict__ W,
                   const float* __restrict__ bias, void* __restrict__ Cout) {
  const int lane = threadIdx.x & 31;
  const int tile = __builtin_amdgcn_readfirstlane(
      blockIdx.x * 8 + (threadIdx.x >> 5));     // 4096 wave tiles
  const int m0 = (tile >> 3) * 32;              // 512 m-slabs of 32 rows
  const int n0 = (tile & 7) * 128;              // 8 n-slabs of 128 cols

  const int lofA = (lane & 15) * EDIM + (lane >> 4) * 8;   // per-lane (A)
  const int lofB = (lane & 15) * EDIM + (lane >> 4) * 8;   // per-lane (B)

  // uniform base pointers pinned to SGPRs; loads: saddr + v(lof+kb) + imm
  gbf16* wt[8];
#pragma unroll
  for (int t = 0; t < 8; ++t) wt[t] = uni_bf(W + (size_t)(n0 + t * 16) * EDIM);
  gf32*  af0 = uni_f((const float*)Ain + (size_t)m0 * EDIM);
  gf32*  af1 = af0 + 16 * EDIM;
  gbf16* ab0 = uni_bf((const __bf16*)Ain + (size_t)m0 * EDIM);
  gbf16* ab1 = ab0 + 16 * EDIM;

  v8f acc[2][8] = {};
#pragma unroll 2
  for (int kb = 0; kb < EDIM; kb += 32) {
    v16bf a0, a1;
    if constexpr (A_BF16) {
      a0 = load_a_bf16_g(ab0 + (lofA + kb));
      a1 = load_a_bf16_g(ab1 + (lofA + kb));
    } else {
      a0 = load_a_f32_g(af0 + (lofA + kb));
      a1 = load_a_f32_g(af1 + (lofA + kb));
    }
#pragma unroll
    for (int t = 0; t < 8; ++t) {
      v16bf bt = load_b_bf16_g(wt[t] + (lofB + kb));
      acc[0][t] = wmma_bf16(a0, bt, acc[0][t]);
      acc[1][t] = wmma_bf16(a1, bt, acc[1][t]);
    }
  }

  // C layout: VGPR e -> M = mtile + e + 8*(lane>=16), N = ntile + (lane&15)
#pragma unroll
  for (int mi = 0; mi < 2; ++mi) {
    const int mbase = m0 + mi * 16 + 8 * (lane >> 4);
#pragma unroll
    for (int t = 0; t < 8; ++t) {
      const int   ncol = n0 + t * 16 + (lane & 15);
      const float bv   = bias[ncol];
#pragma unroll
      for (int e = 0; e < 8; ++e) {
        float v = acc[mi][t][e] + bv;
        int   m = mbase + e;
        if constexpr (OUT_MODE == OUT_ROW_BF16) {
          ((__bf16*)Cout)[(size_t)m * EDIM + ncol] = (__bf16)v;
        } else if constexpr (OUT_MODE == OUT_TRANS_BF16) {
          int b = m >> 12, s = m & (SEQ - 1);
          ((__bf16*)Cout)[((size_t)b * EDIM + ncol) * SEQ + s] = (__bf16)v;
        } else {
          ((float*)Cout)[(size_t)m * EDIM + ncol] = v;
        }
      }
    }
  }
}

// ---------------------------------------------------------------------------
// Kernel 3: banded attention. One wave per 16-row tile (1024 tiles),
// 2 waves / block. Per-wave LDS: 16x273 f32 scores + 16x288 bf16 probs.
// ---------------------------------------------------------------------------
#define TW 273   // score row width (272 band cols + 1 pad vs bank conflicts)
#define PW 288   // prob row width: 272 band cols + 16 zeros, multiple of 32

__global__ __launch_bounds__(64)
void band_attn(const __bf16* __restrict__ Qb, const __bf16* __restrict__ Kb,
               const __bf16* __restrict__ Vt, __bf16* __restrict__ Ctx) {
  __shared__ __align__(16) float  Sld[2][16][TW];
  __shared__ __align__(16) __bf16 Pld[2][16][PW];

  const int lane = threadIdx.x & 31;
  const int w    = threadIdx.x >> 5;
  const int r0 = __builtin_amdgcn_readfirstlane((blockIdx.x * 2 + w) * 16);
  const int b  = r0 >> 12;
  const int s0 = r0 & (SEQ - 1);
  const float scale = 0.03125f;                // 1/sqrt(1024)
  const int lofA = (lane & 15) * EDIM + (lane >> 4) * 8;
  const int lofB = (lane & 15) * EDIM + (lane >> 4) * 8;
  const int rbase = 8 * (lane >> 4);

  // ---- Phase 1: banded scores S = scale * Q K^T into LDS ----
  // Two k-passes of 9 column tiles (tile 8 recomputed identically) to keep
  // acc pressure at 72 VGPRs and let the scheduler buffer loads deeply.
  // A K-tile (j0 mult of 16) is fully in [0,SEQ) or fully out -> uniform
  // clamp of j0 is enough; fully-OOB tiles are masked to -1e9 below.
  {
    gbf16* qbase = uni_bf(Qb + (size_t)r0 * EDIM);
    const __bf16* kbb = Kb + (size_t)b * SEQ * EDIM;

    auto pass = [&](auto ntc, int t0) {
      constexpr int NT = decltype(ntc)::v;
      gbf16* kt[NT];                                  // uniform SGPR bases
#pragma unroll
      for (int t = 0; t < NT; ++t) {
        int j0 = s0 - RAD + 16 * (t0 + t);
        int j0c = j0 < 0 ? 0 : (j0 > SEQ - 16 ? SEQ - 16 : j0);
        kt[t] = uni_bf(kbb + (size_t)j0c * EDIM);
      }
      v8f acc[NT] = {};
#pragma unroll 2
      for (int kb = 0; kb < EDIM; kb += 32) {
        v16bf a = load_a_bf16_g(qbase + (lofA + kb));
#pragma unroll
        for (int t = 0; t < NT; ++t)
          acc[t] = wmma_bf16(a, load_b_bf16_g(kt[t] + (lofB + kb)), acc[t]);
      }
#pragma unroll
      for (int t = 0; t < NT; ++t) {
        const int jr = s0 - RAD + 16 * (t0 + t) + (lane & 15);
        const int c  = 16 * (t0 + t) + (lane & 15);
#pragma unroll
        for (int e = 0; e < 8; ++e) {
          int i = s0 + rbase + e;
          int d = i - jr; if (d < 0) d = -d;
          bool ok = (jr >= 0) && (jr < SEQ) && (d <= RAD);
          Sld[w][rbase + e][c] = ok ? acc[t][e] * scale : -1e9f;
        }
      }
    };
    pass(IC<9>{}, 0);   // tiles 0..8
    pass(IC<9>{}, 8);   // tiles 8..16 (tile 8 rewritten with same values)
  }
  __syncthreads();

  // ---- Phase 2: softmax over the 272-wide band, probs -> bf16 LDS ----
  {
    const int row = lane & 15;
    const int cb  = (lane >> 4) * 136;   // two lanes cooperate per row
    float m = -3.4e38f;
    for (int c = 0; c < 136; ++c) m = fmaxf(m, Sld[w][row][cb + c]);
    m = fmaxf(m, __shfl_xor(m, 16, 32));
    float sum = 0.f;
    for (int c = 0; c < 136; ++c) {
      float e = __expf(Sld[w][row][cb + c] - m);  // -1e9 underflows to 0
      Sld[w][row][cb + c] = e;
      sum += e;
    }
    sum += __shfl_xor(sum, 16, 32);
    const float inv = 1.f / sum;
    for (int c = 0; c < 136; ++c)
      Pld[w][row][cb + c] = (__bf16)(Sld[w][row][cb + c] * inv);
    for (int idx = lane; idx < 16 * 16; idx += 32)   // zero pad cols 272..287
      Pld[w][idx >> 4][272 + (idx & 15)] = (__bf16)0.f;
  }
  __syncthreads();

  // ---- Phase 3: Ctx = P @ V_band ; Vt layout [B][E][S] (transposed) ----
  // Blocked by 8 n-tiles: one P fragment feeds 8 WMMAs.
  const __bf16* prow = &Pld[w][lane & 15][(lane >> 4) * 8];
  const int     lofV = (lane & 15) * SEQ + (lane >> 4) * 8;   // per-lane
#pragma unroll
  for (int ntb = 0; ntb < 8; ++ntb) {       // 8 blocks x 8 n-tiles = 1024 cols
    gbf16* vt8[8];                                            // uniform bases
#pragma unroll
    for (int nt = 0; nt < 8; ++nt)
      vt8[nt] = uni_bf(Vt + (long)b * EDIM * SEQ
                          + (long)(ntb * 128 + nt * 16) * SEQ + (long)(s0 - RAD));
    v8f acc[8] = {};
#pragma unroll
    for (int jc = 0; jc < 9; ++jc) {
      v16bf a = load_a_bf16(prow + jc * 32);
#pragma unroll
      for (int nt = 0; nt < 8; ++nt) {
        // Slightly OOB j reads hit adjacent (finite) ws data; P there is 0.
        acc[nt] = wmma_bf16(a, load_b_bf16_g(vt8[nt] + (lofV + jc * 32)), acc[nt]);
      }
    }
#pragma unroll
    for (int nt = 0; nt < 8; ++nt) {
      const int ncol = ntb * 128 + nt * 16 + (lane & 15);
#pragma unroll
      for (int e = 0; e < 8; ++e)
        Ctx[(size_t)(r0 + rbase + e) * EDIM + ncol] = (__bf16)acc[nt][e];
    }
  }
}

// ---------------------------------------------------------------------------
extern "C" void kernel_launch(void* const* d_in, const int* in_sizes, int n_in,
                              void* d_out, int out_size, void* d_ws, size_t ws_size,
                              hipStream_t stream) {
  (void)in_sizes; (void)n_in; (void)out_size; (void)ws_size;
  const float* query = (const float*)d_in[0];
  const float* key   = (const float*)d_in[1];
  const float* value = (const float*)d_in[2];
  const float* Wq = (const float*)d_in[3]; const float* bq = (const float*)d_in[4];
  const float* Wk = (const float*)d_in[5]; const float* bk = (const float*)d_in[6];
  const float* Wv = (const float*)d_in[7]; const float* bv = (const float*)d_in[8];
  const float* Wo = (const float*)d_in[9]; const float* bo = (const float*)d_in[10];

  const size_t WSZ = (size_t)EDIM * EDIM;   // 1M elems
  const size_t MSZ = (size_t)MROWS * EDIM;  // 16M elems
  __bf16* Wqb = (__bf16*)d_ws;
  __bf16* Wkb = Wqb + WSZ;
  __bf16* Wvb = Wkb + WSZ;
  __bf16* Wob = Wvb + WSZ;
  __bf16* Qb  = Wob + WSZ;
  __bf16* Kb  = Qb + MSZ;
  __bf16* Vt  = Kb + MSZ;
  __bf16* Ctx = Vt + MSZ;   // total ws use: 136 MB

  const int nW = (int)WSZ;
  const int cb = (nW + 255) / 256;
  cvt_f32_bf16<<<cb, 256, 0, stream>>>(Wq, Wqb, nW);
  cvt_f32_bf16<<<cb, 256, 0, stream>>>(Wk, Wkb, nW);
  cvt_f32_bf16<<<cb, 256, 0, stream>>>(Wv, Wvb, nW);
  cvt_f32_bf16<<<cb, 256, 0, stream>>>(Wo, Wob, nW);

  // 4096 wave-tiles (32x128 each) / 8 waves per block = 512 blocks
  gemm32x128_nt<false, OUT_ROW_BF16 ><<<512, 256, 0, stream>>>(query, Wqb, bq, Qb);
  gemm32x128_nt<false, OUT_ROW_BF16 ><<<512, 256, 0, stream>>>(key,   Wkb, bk, Kb);
  gemm32x128_nt<false, OUT_TRANS_BF16><<<512, 256, 0, stream>>>(value, Wvb, bv, Vt);

  band_attn<<<512, 64, 0, stream>>>(Qb, Kb, Vt, Ctx);

  gemm32x128_nt<true, OUT_ROW_F32><<<512, 256, 0, stream>>>(Ctx, Wob, bo, d_out);
}
